// MultiHeadAttention_2190433321584
// MI455X (gfx1250) — matrix operand order repaired and checked by reading, not compiled, not run on hardware
//
#include <hip/hip_runtime.h>
#include <hip/hip_bf16.h>

#define B_  4
#define S_  2048
#define D_  2048
#define H_  16
#define HD_ 128

typedef __bf16 bf16;
typedef __attribute__((ext_vector_type(16))) __bf16 v16bf;
typedef __attribute__((ext_vector_type(8)))  float  v8f;

union FragBits { v16bf v; uint4 u[2]; };

// A-matrix fragment, 16x32 bf16 (ISA 7.12.2): lane m = lane&15, half = lane>>4.
// VGPR0..3 hold K = half*8 .. half*8+7 (16B contiguous), VGPR4..7 hold K = 16+half*8 ..
__device__ __forceinline__ v16bf load_frag_a(const bf16* base, int ld) {
  const int lane = threadIdx.x & 31;
  const int m = lane & 15, lh = lane >> 4;
  const bf16* p = base + m * ld + lh * 8;
  FragBits r;
  r.u[0] = *(const uint4*)(p);
  r.u[1] = *(const uint4*)(p + 16);
  return r.v;
}

// B-matrix fragment, 32x16 bf16, sourced from a row-major (N,K) matrix (i.e. B = Wᵀ):
// lane n = lane&15 picks row n of W; VGPR0..3 hold K = half*8 .. half*8+7, VGPR4..7 hold K = 16+half*8 ..
__device__ __forceinline__ v16bf load_frag_b(const bf16* base, int ld) {
  const int lane = threadIdx.x & 31;
  const int n = lane & 15, lh = lane >> 4;
  const bf16* p = base + n * ld + lh * 8;
  FragBits r;
  r.u[0] = *(const uint4*)(p);
  r.u[1] = *(const uint4*)(p + 16);
  return r.v;
}

__device__ __forceinline__ v8f wmma_bf16(v16bf a, v16bf b, v8f c) {
  return __builtin_amdgcn_wmma_f32_16x16x32_bf16(false, a, false, b, (short)0, c,
                                                 false, false);
}

// Async copy of one 16B chunk global->LDS, tracked on ASYNCcnt.
// LDS offset = low 32 bits of the generic pointer (ISA: LDS aperture addr[31:0]).
__device__ __forceinline__ void async_copy_b128(void* lds_ptr, const void* gptr) {
  unsigned lds_off = (unsigned)(unsigned long long)lds_ptr;
  unsigned long long ga = (unsigned long long)gptr;
  asm volatile("global_load_async_to_lds_b128 %0, %1, off"
               :: "v"(lds_off), "v"(ga) : "memory");
}

__device__ __forceinline__ void wait_asynccnt_8() {
  asm volatile("s_wait_asynccnt 0x8" ::: "memory");
}
__device__ __forceinline__ void wait_asynccnt_0() {
  asm volatile("s_wait_asynccnt 0x0" ::: "memory");
}

// ---------------------------------------------------------------------------
// f32 -> bf16 elementwise convert (grid-stride)
// ---------------------------------------------------------------------------
__global__ void f32_to_bf16(const float* __restrict__ in, bf16* __restrict__ out, int n) {
  int i = blockIdx.x * blockDim.x + threadIdx.x;
  const int stride = gridDim.x * blockDim.x;
  for (; i < n; i += stride) out[i] = (bf16)in[i];
}

// ---------------------------------------------------------------------------
// C(M,N) = A(M,K) * W(N,K)^T  [+ bias], bf16 inputs, f32 output.
// 128 threads = 4 waves arranged 2x2, each wave owns a 64x64 tile.
// A/B K-slices (128x32 each) are staged into LDS with double-buffered
// GLOBAL_LOAD_ASYNC_TO_LDS_B128, drained via s_wait_asynccnt.
// LDS row pitch = 40 elements (80B) so fragment reads are bank-conflict-free.
// ---------------------------------------------------------------------------
#define GPAD 40

__global__ void __launch_bounds__(128) gemm_bf16_nt(
    const bf16* __restrict__ A, const bf16* __restrict__ W,
    float* __restrict__ C, const float* __restrict__ bias,
    int M, int N, int K)
{
  __shared__ __align__(16) bf16 As[2][128 * GPAD];
  __shared__ __align__(16) bf16 Bs[2][128 * GPAD];

  const int tid  = threadIdx.x;
  const int w    = tid >> 5;
  const int wm   = (w >> 1) * 64;           // wave row offset inside block tile
  const int wn   = (w & 1) * 64;            // wave col offset inside block tile
  const int rowB = blockIdx.y * 128;
  const int colB = blockIdx.x * 128;
  const int lane = tid & 31;
  const int nn = lane & 15, lh = lane >> 4;

  const bf16* Arow = A + (size_t)(rowB + tid) * K;  // thread stages one row of each tile
  const bf16* Brow = W + (size_t)(colB + tid) * K;

  // Stage the 32-wide K-slice starting at k0 into buffer `buf` (8 async ops/thread).
  auto stage = [&](int buf, int k0) {
    bf16* la = &As[buf][tid * GPAD];
    bf16* lb = &Bs[buf][tid * GPAD];
    const bf16* ga = Arow + k0;
    const bf16* gb = Brow + k0;
#pragma unroll
    for (int c = 0; c < 4; ++c) {
      async_copy_b128(la + c * 8, ga + c * 8);
      async_copy_b128(lb + c * 8, gb + c * 8);
    }
  };

  v8f acc[4][4];
#pragma unroll
  for (int i = 0; i < 4; ++i)
#pragma unroll
    for (int j = 0; j < 4; ++j) acc[i][j] = {};

  stage(0, 0);

  int buf = 0;
  for (int k0 = 0; k0 < K; k0 += 32) {
    if (k0 + 32 < K) {
      stage(buf ^ 1, k0 + 32);   // prefetch next slice into the other buffer
      wait_asynccnt_8();         // drain this buffer's 8 ops; keep the new 8 in flight
    } else {
      wait_asynccnt_0();
    }
    __syncthreads();

    v16bf af[4], bfr[4];
#pragma unroll
    for (int i = 0; i < 4; ++i)
      af[i] = load_frag_a(&As[buf][(wm + 16 * i) * GPAD], GPAD);
#pragma unroll
    for (int j = 0; j < 4; ++j)
      bfr[j] = load_frag_b(&Bs[buf][(wn + 16 * j) * GPAD], GPAD);
#pragma unroll
    for (int i = 0; i < 4; ++i)
#pragma unroll
      for (int j = 0; j < 4; ++j)
        acc[i][j] = wmma_bf16(af[i], bfr[j], acc[i][j]);

    __syncthreads();             // everyone done reading before buffer is re-staged
    buf ^= 1;
  }

  // C/D layout: lanes 0-15 -> M=r, lanes 16-31 -> M=r+8; N = lane&15.
#pragma unroll
  for (int i = 0; i < 4; ++i)
#pragma unroll
    for (int j = 0; j < 4; ++j)
#pragma unroll
      for (int r = 0; r < 8; ++r) {
        const int row = rowB + wm + 16 * i + r + 8 * lh;
        const int col = colB + wn + 16 * j + nn;
        float v = acc[i][j][r];
        if (bias) v += bias[col];
        C[(size_t)row * N + col] = v;
      }
}

// ---------------------------------------------------------------------------
// RoPE epilogue + layout shuffle. F is the f32 GEMM output laid out (B,S,H,HD).
// mode 0: apply RoPE, emit bf16 (B,H,S,HD)   (Q and K)
// mode 1: no RoPE, emit bf16 (B,H,HD,S)      (V transposed for PV B-fragments)
// ---------------------------------------------------------------------------
__global__ void rope_pack(const float* __restrict__ F, bf16* __restrict__ out, int mode) {
  int idx = blockIdx.x * blockDim.x + threadIdx.x;  // B*S*H*(HD/2) = 8388608
  if (idx >= B_ * S_ * H_ * (HD_ / 2)) return;
  const int i = idx & 63;  idx >>= 6;
  const int h = idx & 15;  idx >>= 4;
  const int s = idx & 2047;
  const int b = idx >> 11;

  const float* p = F + ((size_t)(b * S_ + s) * D_ + h * HD_ + 2 * i);
  const float t1 = p[0], t2 = p[1];

  if (mode == 0) {
    const float inv = __powf(10000.0f, -(float)(2 * i) * (1.0f / 128.0f));
    const float ang = (float)s * inv;
    const float c = __cosf(ang), sn = __sinf(ang);
    bf16* o = out + ((size_t)((b * H_ + h) * S_ + s) * HD_ + 2 * i);
    o[0] = (bf16)(t1 * c - t2 * sn);
    o[1] = (bf16)(t1 * sn + t2 * c);
  } else {
    bf16* o = out + ((size_t)((b * H_ + h) * HD_ + 2 * i) * S_ + s);
    o[0]  = (bf16)t1;
    o[S_] = (bf16)t2;
  }
}

// ---------------------------------------------------------------------------
// Causal flash attention, one wave per 32-query tile per (b,h).
// Q,K: (B,H,S,HD) bf16.  Vt: (B,H,HD,S) bf16.  Ctx out: (B,S,D) bf16.
// ---------------------------------------------------------------------------
__global__ void __launch_bounds__(32) flash_attn(
    const bf16* __restrict__ Q, const bf16* __restrict__ Kmat,
    const bf16* __restrict__ Vt, bf16* __restrict__ Ctx)
{
  __shared__ __align__(16) bf16 Pl[32 * 32];

  const int bh = blockIdx.y;
  const int b  = bh >> 4;
  const int h  = bh & 15;
  const int q0 = blockIdx.x * 32;
  const int lane = threadIdx.x & 31;
  const int nn = lane & 15, lh = lane >> 4;

  const bf16* Qb = Q    + (size_t)bh * S_ * HD_;
  const bf16* Kb = Kmat + (size_t)bh * S_ * HD_;
  const bf16* Vb = Vt   + (size_t)bh * HD_ * S_;

  // Hoist Q fragments: loop-invariant across all key blocks.
  v16bf qf[2][4];
#pragma unroll
  for (int mt = 0; mt < 2; ++mt)
#pragma unroll
    for (int kt = 0; kt < 4; ++kt)
      qf[mt][kt] = load_frag_a(Qb + (size_t)(q0 + 16 * mt) * HD_ + kt * 32, HD_);

  v8f acc[2][8];
#pragma unroll
  for (int mt = 0; mt < 2; ++mt)
#pragma unroll
    for (int j = 0; j < 8; ++j) acc[mt][j] = {};

  float mrow[2][8], lrow[2][8];
#pragma unroll
  for (int mt = 0; mt < 2; ++mt)
#pragma unroll
    for (int r = 0; r < 8; ++r) { mrow[mt][r] = -1e38f; lrow[mt][r] = 0.0f; }

  const float scale = 0.08838834764831845f;  // 1/sqrt(128)
  const int kbmax = q0 >> 5;

  for (int kb = 0; kb <= kbmax; ++kb) {
    const int key0 = kb * 32;

    // ---- scores: S = Q * K^T over HD=128 (4 WMMA K-steps) ----
    v8f sc[2][2];
#pragma unroll
    for (int nt = 0; nt < 2; ++nt) {
      v16bf kf[4];
#pragma unroll
      for (int kt = 0; kt < 4; ++kt)
        kf[kt] = load_frag_b(Kb + (size_t)(key0 + 16 * nt) * HD_ + kt * 32, HD_);
#pragma unroll
      for (int mt = 0; mt < 2; ++mt) {
        v8f s0 = {};
#pragma unroll
        for (int kt = 0; kt < 4; ++kt)
          s0 = wmma_bf16(qf[mt][kt], kf[kt], s0);
        sc[mt][nt] = s0;
      }
    }

    // ---- scale + causal mask (only the diagonal key block needs masking) ----
#pragma unroll
    for (int mt = 0; mt < 2; ++mt)
#pragma unroll
      for (int nt = 0; nt < 2; ++nt)
#pragma unroll
        for (int r = 0; r < 8; ++r) {
          float v = sc[mt][nt][r] * scale;
          if (kb == kbmax) {
            const int key  = key0 + nt * 16 + nn;
            const int qrow = q0 + mt * 16 + r + 8 * lh;
            if (key > qrow) v = -1e30f;
          }
          sc[mt][nt][r] = v;
        }

    // ---- online softmax: rows live across a 16-lane group for fixed VGPR ----
#pragma unroll
    for (int mt = 0; mt < 2; ++mt) {
      float pm[8];
#pragma unroll
      for (int r = 0; r < 8; ++r) pm[r] = fmaxf(sc[mt][0][r], sc[mt][1][r]);
#pragma unroll
      for (int off = 1; off < 16; off <<= 1)
#pragma unroll
        for (int r = 0; r < 8; ++r)
          pm[r] = fmaxf(pm[r], __shfl_xor(pm[r], off, 16));
#pragma unroll
      for (int r = 0; r < 8; ++r) {
        const float mnew = fmaxf(mrow[mt][r], pm[r]);
        const float corr = __expf(mrow[mt][r] - mnew);
        mrow[mt][r] = mnew;
        lrow[mt][r] = lrow[mt][r] * corr;
#pragma unroll
        for (int j = 0; j < 8; ++j) acc[mt][j][r] = acc[mt][j][r] * corr;
        sc[mt][0][r] = __expf(sc[mt][0][r] - mnew);
        sc[mt][1][r] = __expf(sc[mt][1][r] - mnew);
      }
      float ps[8];
#pragma unroll
      for (int r = 0; r < 8; ++r) ps[r] = sc[mt][0][r] + sc[mt][1][r];
#pragma unroll
      for (int off = 1; off < 16; off <<= 1)
#pragma unroll
        for (int r = 0; r < 8; ++r) ps[r] += __shfl_xor(ps[r], off, 16);
#pragma unroll
      for (int r = 0; r < 8; ++r) lrow[mt][r] += ps[r];
    }

    // ---- stage P (C/D layout) into LDS as a 32x32 row-major bf16 tile ----
#pragma unroll
    for (int mt = 0; mt < 2; ++mt)
#pragma unroll
      for (int nt = 0; nt < 2; ++nt)
#pragma unroll
        for (int r = 0; r < 8; ++r)
          Pl[(mt * 16 + r + 8 * lh) * 32 + nt * 16 + nn] = (bf16)sc[mt][nt][r];
    __syncthreads();

    // ---- ctx += P * V : A-fragments from LDS, B-fragments from Vt rows ----
#pragma unroll
    for (int mt = 0; mt < 2; ++mt) {
      v16bf pf = load_frag_a(&Pl[mt * 16 * 32], 32);
#pragma unroll
      for (int j = 0; j < 8; ++j) {
        v16bf vf = load_frag_b(Vb + (size_t)(j * 16) * S_ + key0, S_);
        acc[mt][j] = wmma_bf16(pf, vf, acc[mt][j]);
      }
    }
    __syncthreads();
  }

  // ---- normalize and write ctx back as (B,S,D) bf16 for the output GEMM ----
#pragma unroll
  for (int mt = 0; mt < 2; ++mt)
#pragma unroll
    for (int r = 0; r < 8; ++r) {
      const float inv = 1.0f / lrow[mt][r];
      const int row = q0 + 16 * mt + r + 8 * lh;
#pragma unroll
      for (int j = 0; j < 8; ++j) {
        const int col = j * 16 + nn;
        Ctx[((size_t)(b * S_ + row)) * D_ + h * HD_ + col] = (bf16)(acc[mt][j][r] * inv);
      }
    }
}

// ---------------------------------------------------------------------------
extern "C" void kernel_launch(void* const* d_in, const int* in_sizes, int n_in,
                              void* d_out, int out_size, void* d_ws, size_t ws_size,
                              hipStream_t stream) {
  (void)in_sizes; (void)n_in; (void)out_size; (void)ws_size;

  const float* x  = (const float*)d_in[0];
  const float* wq = (const float*)d_in[1];
  const float* wk = (const float*)d_in[2];
  const float* wv = (const float*)d_in[3];
  const float* wo = (const float*)d_in[4];
  const float* bo = (const float*)d_in[5];
  float* out = (float*)d_out;

  const size_t MROWS = (size_t)B_ * S_;   // 8192
  const size_t XE = MROWS * D_;           // 16,777,216 elements
  const size_t WE = (size_t)D_ * D_;      // 4,194,304 elements

  char* ws = (char*)d_ws;
  bf16*  Xh  = (bf16*)ws;  ws += XE * sizeof(bf16);
  bf16*  Wqh = (bf16*)ws;  ws += WE * sizeof(bf16);
  bf16*  Wkh = (bf16*)ws;  ws += WE * sizeof(bf16);
  bf16*  Wvh = (bf16*)ws;  ws += WE * sizeof(bf16);
  bf16*  Woh = (bf16*)ws;  ws += WE * sizeof(bf16);
  float* Ff  = (float*)ws; ws += XE * sizeof(float);
  bf16*  Qh  = (bf16*)ws;  ws += XE * sizeof(bf16);
  bf16*  Kh  = (bf16*)ws;  ws += XE * sizeof(bf16);
  bf16*  Vt  = (bf16*)ws;  ws += XE * sizeof(bf16);
  bf16*  Ch  = (bf16*)ws;  ws += XE * sizeof(bf16);

  // 1) precision conversion
  f32_to_bf16<<<dim3(4096), dim3(256), 0, stream>>>(x,  Xh,  (int)XE);
  f32_to_bf16<<<dim3(2048), dim3(256), 0, stream>>>(wq, Wqh, (int)WE);
  f32_to_bf16<<<dim3(2048), dim3(256), 0, stream>>>(wk, Wkh, (int)WE);
  f32_to_bf16<<<dim3(2048), dim3(256), 0, stream>>>(wv, Wvh, (int)WE);
  f32_to_bf16<<<dim3(2048), dim3(256), 0, stream>>>(wo, Woh, (int)WE);

  const dim3 gblk(128);
  const dim3 ggrid(D_ / 128, (unsigned)(MROWS / 128));  // (16, 64)
  const dim3 rblk(256);
  const dim3 rgrid((B_ * S_ * H_ * (HD_ / 2)) / 256);

  // 2) Q = X Wq^T, RoPE -> (B,H,S,HD)
  gemm_bf16_nt<<<ggrid, gblk, 0, stream>>>(Xh, Wqh, Ff, nullptr, (int)MROWS, D_, D_);
  rope_pack<<<rgrid, rblk, 0, stream>>>(Ff, Qh, 0);
  // 3) K = X Wk^T, RoPE -> (B,H,S,HD)
  gemm_bf16_nt<<<ggrid, gblk, 0, stream>>>(Xh, Wkh, Ff, nullptr, (int)MROWS, D_, D_);
  rope_pack<<<rgrid, rblk, 0, stream>>>(Ff, Kh, 0);
  // 4) V = X Wv^T -> transposed (B,H,HD,S)
  gemm_bf16_nt<<<ggrid, gblk, 0, stream>>>(Xh, Wvh, Ff, nullptr, (int)MROWS, D_, D_);
  rope_pack<<<rgrid, rblk, 0, stream>>>(Ff, Vt, 1);

  // 5) causal flash attention -> ctx (B,S,D) bf16
  flash_attn<<<dim3(S_ / 32, B_ * H_), dim3(32), 0, stream>>>(Qh, Kh, Vt, Ch);

  // 6) out = ctx Wo^T + bo  (f32 output)
  gemm_bf16_nt<<<ggrid, gblk, 0, stream>>>(Ch, Woh, out, bo, (int)MROWS, D_, D_);
}
